// MultiHeadAttention_82841329205799
// MI455X (gfx1250) — matrix operand order repaired and checked by reading, not compiled, not run on hardware
//
#include <hip/hip_runtime.h>

// ---------------------------------------------------------------------------
// Transformer cross-attention block for MI455X (gfx1250, wave32, WMMA).
// All contractions via v_wmma_f32_16x16x32_bf16 (f32 accumulate).
// Weights are pre-transposed to [N,K] so every WMMA operand fragment is two
// contiguous global_load_b128 per lane (no scalar b16 loads / VALU packing).
// ---------------------------------------------------------------------------

typedef __attribute__((ext_vector_type(16))) __bf16   v16bf;
typedef __attribute__((ext_vector_type(8)))  float    v8f;
typedef __attribute__((ext_vector_type(8)))  unsigned v8u;
typedef __attribute__((ext_vector_type(4)))  unsigned v4u;

static constexpr int B_  = 8;
static constexpr int N_  = 1024;   // NX == NY
static constexpr int C_  = 1024;   // C1 == C2
static constexpr int H_  = 16;
static constexpr int D_  = 32;
static constexpr int HD_ = H_ * D_;        // 512
static constexpr int M_  = B_ * N_;        // 8192 rows
static constexpr int WC_ = 4 * C_;         // 4096

static __device__ __forceinline__ v8f wmma_bf16(v16bf a, v16bf b, v8f c) {
  return __builtin_amdgcn_wmma_f32_16x16x32_bf16(
      /*neg_a=*/false, a, /*neg_b=*/false, b,
      /*c_mod=*/(short)0, c, /*reuse_a=*/false, /*reuse_b=*/false);
}

// A-matrix 16x32 bf16 tile (ISA 7.12.2): lane<16 row M=lane holds K 0..7 and
// 16..23; lane>=16 same row holds K 8..15 and 24..31. Two b128 loads per lane.
static __device__ __forceinline__ v16bf load_A16(const __bf16* A, int lda,
                                                 int row0, int k0, int lane) {
  const int r = lane & 15, hi = lane >> 4;
  const __bf16* base = A + (size_t)(row0 + r) * lda + k0 + hi * 8;
  v4u lo = *(const v4u*)base;
  v4u hv = *(const v4u*)(base + 16);
  v8u u;
  u[0] = lo[0]; u[1] = lo[1]; u[2] = lo[2]; u[3] = lo[3];
  u[4] = hv[0]; u[5] = hv[1]; u[6] = hv[2]; u[7] = hv[3];
  return __builtin_bit_cast(v16bf, u);
}

// B-matrix 32x16 bf16 tile from a TRANSPOSED [N,K] source (element(K,N) =
// S[N*lds + K]): lane<16 column N=lane holds K 0..7 and 16..23, lane>=16 holds K 8..15 and 24..31.
// Two 8-element runs per lane -> two b128 loads.
static __device__ __forceinline__ v16bf load_Bt(const __bf16* S, int lds,
                                                int k0, int col0, int lane) {
  const int c = lane & 15, hi = lane >> 4;
  const __bf16* base = S + (size_t)(col0 + c) * lds + k0 + hi * 8;
  v4u lo = *(const v4u*)base;
  v4u hv = *(const v4u*)(base + 16);
  v8u u;
  u[0] = lo[0]; u[1] = lo[1]; u[2] = lo[2]; u[3] = lo[3];
  u[4] = hv[0]; u[5] = hv[1]; u[6] = hv[2]; u[7] = hv[3];
  return __builtin_bit_cast(v16bf, u);
}

// ---------------------------------------------------------------------------
// fp32 [K,N] -> bf16 [N,K] transpose-convert (weights; one-time cost).
// grid = (K/32, N/8, Z) with per-z dense K*N slabs (stacked per-head weights).
// ---------------------------------------------------------------------------
__global__ void k_cvt_tr(const float* __restrict__ s, __bf16* __restrict__ d,
                         int K, int N) {
  const size_t zoff = (size_t)blockIdx.z * K * N;
  const int k = blockIdx.x * 32 + (threadIdx.x & 31);
  const int n = blockIdx.y * 8 + (threadIdx.x >> 5);
  if (k < K && n < N)
    d[zoff + (size_t)n * K + k] = (__bf16)s[zoff + (size_t)k * N + n];
}

// ---------------------------------------------------------------------------
// LayerNorm over rows of [R, 1024]; writes optional fp32 copy + bf16 copy.
// ---------------------------------------------------------------------------
__global__ __launch_bounds__(256) void k_ln(const float* __restrict__ X,
                                            const float* __restrict__ g,
                                            const float* __restrict__ be,
                                            float* __restrict__ outf,
                                            __bf16* __restrict__ outb) {
  const int row = blockIdx.x, t = threadIdx.x;
  const float* xr = X + (size_t)row * C_;
  __shared__ float sh[256];
  float x[4], s = 0.f;
#pragma unroll
  for (int j = 0; j < 4; ++j) { x[j] = xr[t + 256 * j]; s += x[j]; }
  sh[t] = s; __syncthreads();
  for (int st = 128; st > 0; st >>= 1) { if (t < st) sh[t] += sh[t + st]; __syncthreads(); }
  const float mu = sh[0] * (1.f / C_);
  __syncthreads();
  float vs = 0.f;
#pragma unroll
  for (int j = 0; j < 4; ++j) { float d = x[j] - mu; vs += d * d; }
  sh[t] = vs; __syncthreads();
  for (int st = 128; st > 0; st >>= 1) { if (t < st) sh[t] += sh[t + st]; __syncthreads(); }
  const float rs = rsqrtf(sh[0] * (1.f / C_) + 1e-5f);
#pragma unroll
  for (int j = 0; j < 4; ++j) {
    const int cidx = t + 256 * j;
    const float val = (x[j] - mu) * rs * g[cidx] + be[cidx];
    if (outf) outf[(size_t)row * C_ + cidx] = val;
    outb[(size_t)row * C_ + cidx] = (__bf16)val;
  }
}

// ---------------------------------------------------------------------------
// QKV projections. Per head h: [8192 x 1024] x [1024 x 32].
// Wave tile = 16 rows x 32 cols (A reused for both column tiles).
// q,k stored [H][M][32] row-major; v stored transposed [H][B][32][1024] so the
// attention P@V operand is contiguous.
// grid = (M/128, H, 3), 8 waves/block.
// ---------------------------------------------------------------------------
__global__ __launch_bounds__(256) void k_qkv(const __bf16* __restrict__ xn,
                                             const __bf16* __restrict__ yn,
                                             const __bf16* __restrict__ Wqt,
                                             const __bf16* __restrict__ Wkt,
                                             const __bf16* __restrict__ Wvt,
                                             __bf16* __restrict__ q,
                                             __bf16* __restrict__ k,
                                             __bf16* __restrict__ v) {
  const int h = blockIdx.y, z = blockIdx.z;
  const int wave = threadIdx.x >> 5, lane = threadIdx.x & 31;
  const __bf16* A  = (z == 0) ? xn : yn;
  const __bf16* Wt = ((z == 0) ? Wqt : (z == 1) ? Wkt : Wvt) + (size_t)h * D_ * C_;
  const int row0 = blockIdx.x * 128 + wave * 16;
  v8f acc0 = {}, acc1 = {};
  for (int k0 = 0; k0 < C_; k0 += 32) {
    __builtin_prefetch(A + (size_t)row0 * C_ + k0 + 128, 0, 1);
    v16bf a  = load_A16(A, C_, row0, k0, lane);
    v16bf b0 = load_Bt(Wt, C_, k0, 0, lane);
    v16bf b1 = load_Bt(Wt, C_, k0, 16, lane);
    acc0 = wmma_bf16(a, b0, acc0);
    acc1 = wmma_bf16(a, b1, acc1);
  }
  const int hi = lane >> 4, c = lane & 15;
  if (z == 2) {
    // v^T: [H][B][D][N]
#pragma unroll
    for (int i = 0; i < 8; ++i) {
      const int r = row0 + i + 8 * hi;
      const int bb = r >> 10, n = r & (N_ - 1);
      __bf16* vt = v + (size_t)h * M_ * D_ + (size_t)bb * D_ * N_;
      vt[(size_t)c * N_ + n]        = (__bf16)acc0[i];
      vt[(size_t)(16 + c) * N_ + n] = (__bf16)acc1[i];
    }
  } else {
    __bf16* O = ((z == 0) ? q : k) + (size_t)h * M_ * D_;
#pragma unroll
    for (int i = 0; i < 8; ++i) {
      const int r = row0 + i + 8 * hi;
      O[(size_t)r * D_ + c]      = (__bf16)acc0[i];
      O[(size_t)r * D_ + 16 + c] = (__bf16)acc1[i];
    }
  }
}

// ---------------------------------------------------------------------------
// Flash attention: grid = (N/128, H, B); each wave owns 16 query rows,
// streams keys/values in 32-wide chunks with online softmax.
// S = q(16x32) x k^T(32x16): k is [M][32] so kT fragments are contiguous.
// O += P(16x32) x V(32x32): V is pre-transposed, fragments contiguous.
// P relaid into A-matrix layout through per-wave LDS (same-wave DS in-order).
// ---------------------------------------------------------------------------
__global__ __launch_bounds__(256) void k_attn(const __bf16* __restrict__ q,
                                              const __bf16* __restrict__ k,
                                              const __bf16* __restrict__ v,
                                              __bf16* __restrict__ o) {
  const int b = blockIdx.z, h = blockIdx.y;
  const int wave = threadIdx.x >> 5, lane = threadIdx.x & 31;
  const int hi = lane >> 4, c = lane & 15;
  __shared__ alignas(16) __bf16 Pl[8][16 * 32];

  const __bf16* qp  = q + ((size_t)h * M_ + (size_t)b * N_) * D_;
  const __bf16* kp  = k + ((size_t)h * M_ + (size_t)b * N_) * D_;
  const __bf16* vtp = v + (size_t)h * M_ * D_ + (size_t)b * D_ * N_;

  const int qr0 = blockIdx.x * 128 + wave * 16;
  const v16bf aq = load_A16(qp, D_, qr0, 0, lane);

  v8f o1 = {}, o2 = {};
  float Mr[8], L[8];
#pragma unroll
  for (int i = 0; i < 8; ++i) { Mr[i] = -1e30f; L[i] = 0.f; }
  const float scale = 0.17677669529663687f; // 1/sqrt(D)

  for (int mj = 0; mj < N_; mj += 32) {
    v16bf bk1 = load_Bt(kp, D_, 0, mj, lane);       // k^T tile, cols mj..mj+15
    v16bf bk2 = load_Bt(kp, D_, 0, mj + 16, lane);
    v8f zero = {};
    v8f s1 = wmma_bf16(aq, bk1, zero);
    v8f s2 = wmma_bf16(aq, bk2, zero);
#pragma unroll
    for (int i = 0; i < 8; ++i) {
      const float x1 = s1[i] * scale, x2 = s2[i] * scale;
      float t = fmaxf(x1, x2);
      t = fmaxf(t, __shfl_xor(t, 1, 32));
      t = fmaxf(t, __shfl_xor(t, 2, 32));
      t = fmaxf(t, __shfl_xor(t, 4, 32));
      t = fmaxf(t, __shfl_xor(t, 8, 32));
      const float mn = fmaxf(Mr[i], t);
      const float p1 = expf(x1 - mn), p2 = expf(x2 - mn);
      float rs = p1 + p2;
      rs += __shfl_xor(rs, 1, 32);
      rs += __shfl_xor(rs, 2, 32);
      rs += __shfl_xor(rs, 4, 32);
      rs += __shfl_xor(rs, 8, 32);
      const float al = expf(Mr[i] - mn);
      L[i] = L[i] * al + rs;
      Mr[i] = mn;
      o1[i] *= al; o2[i] *= al;
      const int r = i + 8 * hi;
      Pl[wave][r * 32 + c]      = (__bf16)p1;
      Pl[wave][r * 32 + 16 + c] = (__bf16)p2;
    }
    v16bf pa  = load_A16(&Pl[wave][0], 32, 0, 0, lane);
    v16bf bv1 = load_Bt(vtp, N_, mj, 0, lane);      // V tile cols d=0..15
    v16bf bv2 = load_Bt(vtp, N_, mj, 16, lane);     // d=16..31
    o1 = wmma_bf16(pa, bv1, o1);
    o2 = wmma_bf16(pa, bv2, o2);
  }

#pragma unroll
  for (int i = 0; i < 8; ++i) {
    const int r = qr0 + i + 8 * hi;
    const float inv = 1.f / L[i];
    const size_t base = ((size_t)b * N_ + r) * HD_ + (size_t)h * D_;
    o[base + c]      = (__bf16)(o1[i] * inv);
    o[base + 16 + c] = (__bf16)(o2[i] * inv);
  }
}

// ---------------------------------------------------------------------------
// Generic bf16 WMMA GEMM, B pre-transposed [N,K]. Wave tile 32x64 (8 wmmas /
// K-step, 12 b128 loads), block tile 128x128 over 8 waves.
// Epilogue: bias / exact-GELU / fp32 residual; fp32 and/or bf16 outputs.
// grid = (M/128, N/128).
// ---------------------------------------------------------------------------
template <int GELU>
__global__ __launch_bounds__(256) void k_gemm(const __bf16* __restrict__ A,
                                              const __bf16* __restrict__ Bt,
                                              const float* __restrict__ bias,
                                              const float* __restrict__ resid,
                                              float* __restrict__ outf,
                                              __bf16* __restrict__ outb,
                                              int N, int K) {
  const int wave = threadIdx.x >> 5, lane = threadIdx.x & 31;
  const int row0 = blockIdx.x * 128 + (wave >> 1) * 32;
  const int col0 = blockIdx.y * 128 + (wave & 1) * 64;
  v8f acc[2][4];
#pragma unroll
  for (int rr = 0; rr < 2; ++rr)
#pragma unroll
    for (int cc = 0; cc < 4; ++cc) acc[rr][cc] = (v8f){};

  for (int k0 = 0; k0 < K; k0 += 32) {
    __builtin_prefetch(A + (size_t)row0 * K + k0 + 128, 0, 1);
    __builtin_prefetch(Bt + (size_t)col0 * K + k0 + 128, 0, 1);
    v16bf a0 = load_A16(A, K, row0, k0, lane);
    v16bf a1 = load_A16(A, K, row0 + 16, k0, lane);
    v16bf b0 = load_Bt(Bt, K, k0, col0, lane);
    v16bf b1 = load_Bt(Bt, K, k0, col0 + 16, lane);
    v16bf b2 = load_Bt(Bt, K, k0, col0 + 32, lane);
    v16bf b3 = load_Bt(Bt, K, k0, col0 + 48, lane);
    acc[0][0] = wmma_bf16(a0, b0, acc[0][0]);
    acc[0][1] = wmma_bf16(a0, b1, acc[0][1]);
    acc[0][2] = wmma_bf16(a0, b2, acc[0][2]);
    acc[0][3] = wmma_bf16(a0, b3, acc[0][3]);
    acc[1][0] = wmma_bf16(a1, b0, acc[1][0]);
    acc[1][1] = wmma_bf16(a1, b1, acc[1][1]);
    acc[1][2] = wmma_bf16(a1, b2, acc[1][2]);
    acc[1][3] = wmma_bf16(a1, b3, acc[1][3]);
  }

  const int hi = lane >> 4, c = lane & 15;
#pragma unroll
  for (int cc = 0; cc < 4; ++cc) {
    const int col = col0 + cc * 16 + c;
    const float bv = bias ? bias[col] : 0.f;
#pragma unroll
    for (int rr = 0; rr < 2; ++rr) {
#pragma unroll
      for (int i = 0; i < 8; ++i) {
        const int r = row0 + rr * 16 + i + 8 * hi;
        float val = acc[rr][cc][i] + bv;
        if (GELU) val = 0.5f * val * (1.f + erff(val * 0.70710678118654752f));
        if (resid) val += resid[(size_t)r * N + col];
        if (outf) outf[(size_t)r * N + col] = val;
        if (outb) outb[(size_t)r * N + col] = (__bf16)val;
      }
    }
  }
}

// ---------------------------------------------------------------------------
// Host orchestration
// ---------------------------------------------------------------------------
extern "C" void kernel_launch(void* const* d_in, const int* in_sizes, int n_in,
                              void* d_out, int out_size, void* d_ws, size_t ws_size,
                              hipStream_t stream) {
  (void)in_sizes; (void)n_in; (void)out_size; (void)ws_size;
  const float* x   = (const float*)d_in[0];
  const float* y   = (const float*)d_in[1];
  const float* Wq  = (const float*)d_in[2];
  const float* Wk  = (const float*)d_in[3];
  const float* Wv  = (const float*)d_in[4];
  const float* W1  = (const float*)d_in[5];
  const float* b1  = (const float*)d_in[6];
  const float* g1  = (const float*)d_in[7];
  const float* be1 = (const float*)d_in[8];
  const float* g2  = (const float*)d_in[9];
  const float* be2 = (const float*)d_in[10];
  const float* g3  = (const float*)d_in[11];
  const float* be3 = (const float*)d_in[12];
  const float* W2  = (const float*)d_in[13];
  const float* b2  = (const float*)d_in[14];
  const float* W3  = (const float*)d_in[15];
  const float* b3  = (const float*)d_in[16];
  float* out = (float*)d_out;

  char* ws = (char*)d_ws;
  auto alloc = [&](size_t bytes) -> void* {
    void* p = (void*)ws;
    ws += (bytes + 255) & ~(size_t)255;
    return p;
  };

  float*  xnf  = (float*)alloc((size_t)M_ * C_ * 4);
  __bf16* xnb  = (__bf16*)alloc((size_t)M_ * C_ * 2);
  __bf16* ynb  = (__bf16*)alloc((size_t)M_ * C_ * 2);
  __bf16* qb   = (__bf16*)alloc((size_t)H_ * M_ * D_ * 2);
  __bf16* kb   = (__bf16*)alloc((size_t)H_ * M_ * D_ * 2);
  __bf16* vb   = (__bf16*)alloc((size_t)H_ * M_ * D_ * 2);  // transposed per (h,b)
  __bf16* ob   = (__bf16*)alloc((size_t)M_ * HD_ * 2);
  float*  out0 = (float*)alloc((size_t)M_ * C_ * 4);
  __bf16* ln3b = (__bf16*)alloc((size_t)M_ * C_ * 2);
  __bf16* h1b  = (__bf16*)alloc((size_t)M_ * WC_ * 2);
  __bf16* Wqt  = (__bf16*)alloc((size_t)H_ * D_ * C_ * 2);
  __bf16* Wkt  = (__bf16*)alloc((size_t)H_ * D_ * C_ * 2);
  __bf16* Wvt  = (__bf16*)alloc((size_t)H_ * D_ * C_ * 2);
  __bf16* W1t  = (__bf16*)alloc((size_t)C_ * HD_ * 2);
  __bf16* W2t  = (__bf16*)alloc((size_t)WC_ * C_ * 2);
  __bf16* W3t  = (__bf16*)alloc((size_t)C_ * WC_ * 2);

  // Transpose-convert weights (fp32 [K,N] -> bf16 [N,K]).
  k_cvt_tr<<<dim3(C_ / 32, D_ / 8, H_), 256, 0, stream>>>(Wq, Wqt, C_, D_);
  k_cvt_tr<<<dim3(C_ / 32, D_ / 8, H_), 256, 0, stream>>>(Wk, Wkt, C_, D_);
  k_cvt_tr<<<dim3(C_ / 32, D_ / 8, H_), 256, 0, stream>>>(Wv, Wvt, C_, D_);
  k_cvt_tr<<<dim3(HD_ / 32, C_ / 8, 1), 256, 0, stream>>>(W1, W1t, HD_, C_);
  k_cvt_tr<<<dim3(C_ / 32, WC_ / 8, 1), 256, 0, stream>>>(W2, W2t, C_, WC_);
  k_cvt_tr<<<dim3(WC_ / 32, C_ / 8, 1), 256, 0, stream>>>(W3, W3t, WC_, C_);

  // LayerNorms of x and y.
  k_ln<<<M_, 256, 0, stream>>>(x, g1, be1, xnf, xnb);
  k_ln<<<M_, 256, 0, stream>>>(y, g2, be2, nullptr, ynb);

  // QKV projections (v emitted transposed).
  k_qkv<<<dim3(M_ / 128, H_, 3), 256, 0, stream>>>(xnb, ynb, Wqt, Wkt, Wvt, qb, kb, vb);

  // Attention.
  k_attn<<<dim3(N_ / 128, H_, B_), 256, 0, stream>>>(qb, kb, vb, ob);

  // out0 = xn + o @ W1 + b1   (fp32)
  k_gemm<0><<<dim3(M_ / 128, C_ / 128), 256, 0, stream>>>(
      ob, W1t, b1, xnf, out0, (__bf16*)nullptr, C_, HD_);

  // LN(out0) -> bf16
  k_ln<<<M_, 256, 0, stream>>>(out0, g3, be3, nullptr, ln3b);

  // h1 = gelu(ln3 @ W2 + b2)  (bf16)
  k_gemm<1><<<dim3(M_ / 128, WC_ / 128), 256, 0, stream>>>(
      ln3b, W2t, b2, (const float*)nullptr, (float*)nullptr, h1b, WC_, C_);

  // out = out0 + h1 @ W3 + b3 (fp32, final output)
  k_gemm<0><<<dim3(M_ / 128, C_ / 128), 256, 0, stream>>>(
      h1b, W3t, b3, out0, out, (__bf16*)nullptr, C_, WC_);
}
